// PointerNet_36395552866908
// MI455X (gfx1250) — matrix operand order repaired and checked by reading, not compiled, not run on hardware
//
#include <hip/hip_runtime.h>
#include <cmath>

// Problem dims (fixed by reference)
#define Bb 32
#define Ss 512
#define Dd 256
#define Hh 512
#define HH4 2048   // 4*H

typedef __attribute__((ext_vector_type(8)))  __bf16 v8bf;
typedef __attribute__((ext_vector_type(16))) __bf16 v16bf;
typedef __attribute__((ext_vector_type(8)))  float  v8f;

// ---------------------------------------------------------------------------
// WMMA fragment loaders (wave32; A and B share contiguous k = 16*(lane/16) + s, not the ISA table order)
// A: 16x32 bf16, M across lanes, K=0..15 (lanes 0-15) and
//    16..31 (lanes 16-31), 2 bf16 per VGPR.
// B: 32x16 bf16 taken from W stored [N,K] row-major: lane n holds column n,
//    K=0..15 (lanes 0-15) or 16..31 (lanes 16-31) -> one contiguous 32B load.
// C/D: 8 VGPRs, lanes 0-15 rows 0..7, lanes 16-31 rows 8..15, col = lane&15.
// ---------------------------------------------------------------------------
__device__ __forceinline__ v16bf load_a_frag(const __bf16* __restrict__ A, int lda, int lane) {
  const int m  = lane & 15;
  const int kb = (lane >> 4) << 4;               // 0 or 16
  v8bf lo = *(const v8bf*)(A + (size_t)m * lda + kb);
  v8bf hi = *(const v8bf*)(A + (size_t)m * lda + kb + 8);
  v16bf r;
#pragma unroll
  for (int i = 0; i < 8; ++i) { r[i] = lo[i]; r[8 + i] = hi[i]; }
  return r;
}

__device__ __forceinline__ v16bf load_b_frag(const __bf16* __restrict__ W, int ldw, int lane) {
  const int n  = lane & 15;
  const int kk = (lane >> 4) << 4;               // 0 or 16
  return *(const v16bf*)(W + (size_t)n * ldw + kk);
}

__device__ __forceinline__ v8f wmma_bf16(v16bf a, v16bf b, v8f c) {
  return __builtin_amdgcn_wmma_f32_16x16x32_bf16(false, a, false, b, (short)0, c, false, false);
}

// ---------------------------------------------------------------------------
// C = A[M,K](bf16) * W[N,K]^T(bf16) + bias. One 32x32 tile per wave:
// 2x2 WMMA register blocking (A/B fragments reused twice -> 16 flop/byte from
// cache) with software-pipelined K loop (next fragments in flight while the
// matrix pipe runs). M, N must be multiples of 32.
// ---------------------------------------------------------------------------
template <typename TOut>
__global__ void wmma_gemm_bias(const __bf16* __restrict__ A,
                               const __bf16* __restrict__ W,
                               const float* __restrict__ bias,
                               TOut* __restrict__ C,
                               int M, int N, int K) {
  const int lane = threadIdx.x & 31;
  const int wave = threadIdx.x >> 5;
  const int tile = blockIdx.x * (blockDim.x >> 5) + wave;
  const int ntn  = N >> 5;
  const int tm   = tile / ntn;
  const int tn   = tile - tm * ntn;
  if (tm >= (M >> 5)) return;                    // wave-uniform guard (EXEC all-1 for WMMA)
  const int m0 = tm << 5, n0 = tn << 5;
  const int col = lane & 15;

  v8f acc00, acc01, acc10, acc11;
  {
    const float bv0 = bias ? bias[n0 + col] : 0.0f;
    const float bv1 = bias ? bias[n0 + 16 + col] : 0.0f;
#pragma unroll
    for (int i = 0; i < 8; ++i) { acc00[i] = bv0; acc10[i] = bv0; acc01[i] = bv1; acc11[i] = bv1; }
  }

  const __bf16* Ap0 = A + (size_t)m0 * K;
  const __bf16* Ap1 = A + (size_t)(m0 + 16) * K;
  const __bf16* Wp0 = W + (size_t)n0 * K;
  const __bf16* Wp1 = W + (size_t)(n0 + 16) * K;

  // prologue
  v16bf a0 = load_a_frag(Ap0, K, lane);
  v16bf a1 = load_a_frag(Ap1, K, lane);
  v16bf b0 = load_b_frag(Wp0, K, lane);
  v16bf b1 = load_b_frag(Wp1, K, lane);

  for (int k1 = 32; k1 < K; k1 += 32) {
    v16bf a0n = load_a_frag(Ap0 + k1, K, lane);
    v16bf a1n = load_a_frag(Ap1 + k1, K, lane);
    v16bf b0n = load_b_frag(Wp0 + k1, K, lane);
    v16bf b1n = load_b_frag(Wp1 + k1, K, lane);
    acc00 = wmma_bf16(a0, b0, acc00);
    acc01 = wmma_bf16(a0, b1, acc01);
    acc10 = wmma_bf16(a1, b0, acc10);
    acc11 = wmma_bf16(a1, b1, acc11);
    a0 = a0n; a1 = a1n; b0 = b0n; b1 = b1n;
  }
  // epilogue
  acc00 = wmma_bf16(a0, b0, acc00);
  acc01 = wmma_bf16(a0, b1, acc01);
  acc10 = wmma_bf16(a1, b0, acc10);
  acc11 = wmma_bf16(a1, b1, acc11);

  const int rbase = (lane >> 4) << 3;
#pragma unroll
  for (int r = 0; r < 8; ++r) {
    const size_t row0 = (size_t)(m0 + rbase + r) * N;
    const size_t row1 = (size_t)(m0 + 16 + rbase + r) * N;
    C[row0 + n0 + col]      = (TOut)acc00[r];
    C[row0 + n0 + 16 + col] = (TOut)acc01[r];
    C[row1 + n0 + col]      = (TOut)acc10[r];
    C[row1 + n0 + 16 + col] = (TOut)acc11[r];
  }
}

// ---------------------------------------------------------------------------
// One LSTM timestep, fused: z = hprev @ Whh^T (+ xp | + bias), gate math,
// state update, bf16 re-quantization of h (double buffered), optional
// hs accumulation. Grid: H/16 blocks x 64 threads (2 waves = 2 batch tiles).
// Each wave: 4 WMMA accumulators (one per gate), software-pipelined K loop.
// ---------------------------------------------------------------------------
__global__ void lstm_step(const __bf16* __restrict__ hprev,   // [B,H] bf16
                          const __bf16* __restrict__ Whh,     // [4H,H] bf16
                          const __bf16* __restrict__ xp,      // [B,4H] bf16 (bias folded) or null
                          const float* __restrict__ bias,     // [4H] or null
                          float* __restrict__ h,              // [B,H] fp32
                          float* __restrict__ c,              // [B,H] fp32 (in-place)
                          __bf16* __restrict__ hnext,         // [B,H] bf16
                          float* __restrict__ hs_out,         // [B,H] or null
                          int accumulate) {
  const int lane = threadIdx.x & 31;
  const int mt   = threadIdx.x >> 5;             // batch tile 0/1
  const int j0   = blockIdx.x << 4;              // hidden-unit tile
  const int m0   = mt << 4;
  const int col  = lane & 15;

  v8f acc[4];
#pragma unroll
  for (int g = 0; g < 4; ++g) {
    const float bv = bias ? bias[g * Hh + j0 + col] : 0.0f;
#pragma unroll
    for (int i = 0; i < 8; ++i) acc[g][i] = bv;
  }

  const __bf16* Ap = hprev + (size_t)m0 * Hh;
  const __bf16* Wg[4];
#pragma unroll
  for (int g = 0; g < 4; ++g) Wg[g] = Whh + (size_t)(g * Hh + j0) * Hh;

  // prologue
  v16bf a = load_a_frag(Ap, Hh, lane);
  v16bf bf[4];
#pragma unroll
  for (int g = 0; g < 4; ++g) bf[g] = load_b_frag(Wg[g], Hh, lane);

  for (int k1 = 32; k1 < Hh; k1 += 32) {
    v16bf an = load_a_frag(Ap + k1, Hh, lane);
    v16bf bn[4];
#pragma unroll
    for (int g = 0; g < 4; ++g) bn[g] = load_b_frag(Wg[g] + k1, Hh, lane);
#pragma unroll
    for (int g = 0; g < 4; ++g) acc[g] = wmma_bf16(a, bf[g], acc[g]);
    a = an;
#pragma unroll
    for (int g = 0; g < 4; ++g) bf[g] = bn[g];
  }
#pragma unroll
  for (int g = 0; g < 4; ++g) acc[g] = wmma_bf16(a, bf[g], acc[g]);

  const int rbase = (lane >> 4) << 3;
#pragma unroll
  for (int r = 0; r < 8; ++r) {
    const int bi = m0 + rbase + r;
    const int j  = j0 + col;
    float zi = acc[0][r], zf = acc[1][r], zg = acc[2][r], zo = acc[3][r];
    if (xp) {
      const __bf16* xr = xp + (size_t)bi * HH4;
      zi += (float)xr[0 * Hh + j];
      zf += (float)xr[1 * Hh + j];
      zg += (float)xr[2 * Hh + j];
      zo += (float)xr[3 * Hh + j];
    }
    const size_t idx = (size_t)bi * Hh + j;
    const float si = 1.0f / (1.0f + __expf(-zi));
    const float sf = 1.0f / (1.0f + __expf(-zf));
    const float so = 1.0f / (1.0f + __expf(-zo));
    const float cn = sf * c[idx] + si * tanhf(zg);
    const float hn = so * tanhf(cn);
    c[idx] = cn;
    h[idx] = hn;
    hnext[idx] = (__bf16)hn;
    if (hs_out) hs_out[idx] = accumulate ? (hs_out[idx] + hn) : hn;
  }
}

// ---------------------------------------------------------------------------
// Attention scores for decoder step t:
//   out[b, t, s] = sum_h tanh(pre1[s,b,h] + q[b,h]) * vt[h] + bvt
// One wave per (s,b) pair; float4 loads; pre1 (33.5 MB) stays L2-resident.
// ---------------------------------------------------------------------------
__global__ void attn_scores(const float* __restrict__ pre1,   // [S*B, H]
                            const float* __restrict__ q,      // [B, H]
                            const float* __restrict__ vt,     // [H]
                            const float* __restrict__ bvt,    // [1]
                            float* __restrict__ out,          // [B, S, S]
                            int t) {
  const int lane = threadIdx.x & 31;
  const int wave = threadIdx.x >> 5;
  const int pair = blockIdx.x * (blockDim.x >> 5) + wave;     // s*B + b
  if (pair >= Ss * Bb) return;
  const int b = pair & (Bb - 1);
  const int s = pair >> 5;
  const float4* pr = (const float4*)(pre1 + (size_t)pair * Hh);
  const float4* qr = (const float4*)(q + (size_t)b * Hh);
  const float4* vr = (const float4*)vt;
  float v = 0.0f;
#pragma unroll
  for (int i = lane; i < Hh / 4; i += 32) {
    float4 p = pr[i], qq = qr[i], vv = vr[i];
    v += tanhf(p.x + qq.x) * vv.x;
    v += tanhf(p.y + qq.y) * vv.y;
    v += tanhf(p.z + qq.z) * vv.z;
    v += tanhf(p.w + qq.w) * vv.w;
  }
#pragma unroll
  for (int off = 16; off > 0; off >>= 1)
    v += __shfl_xor(v, off, 32);
  if (lane == 0)
    out[((size_t)b * Ss + t) * Ss + s] = v + bvt[0];
}

// ---------------------------------------------------------------------------
// Small utility kernels
// ---------------------------------------------------------------------------
__global__ void f2bf_kernel(const float* __restrict__ in, __bf16* __restrict__ out, size_t n) {
  size_t i = (size_t)blockIdx.x * blockDim.x + threadIdx.x;
  if (i < n) out[i] = (__bf16)in[i];
}

// x[B,S,D] fp32 -> xbf[(s*B+b), D] bf16 (GEMM row-major A)
__global__ void x_transpose_bf(const float* __restrict__ x, __bf16* __restrict__ out) {
  size_t i = (size_t)blockIdx.x * blockDim.x + threadIdx.x;
  if (i >= (size_t)Ss * Bb * Dd) return;
  const int d  = (int)(i % Dd);
  const int sb = (int)(i / Dd);
  const int b  = sb & (Bb - 1);
  const int s  = sb >> 5;
  out[i] = (__bf16)x[((size_t)b * Ss + s) * Dd + d];
}

__global__ void concat_bf(const float* __restrict__ a, const float* __restrict__ bsrc,
                          __bf16* __restrict__ out) {
  int i = blockIdx.x * blockDim.x + threadIdx.x;              // over B*2H
  if (i >= Bb * 2 * Hh) return;
  const int b = i >> 10;
  const int k = i & 1023;
  const float v = (k < Hh) ? a[b * Hh + k] : bsrc[b * Hh + (k - Hh)];
  out[i] = (__bf16)v;
}

__global__ void zero_kernel(unsigned int* __restrict__ p, size_t nwords) {
  size_t i = (size_t)blockIdx.x * blockDim.x + threadIdx.x;
  if (i < nwords) p[i] = 0u;
}

// ---------------------------------------------------------------------------
extern "C" void kernel_launch(void* const* d_in, const int* in_sizes, int n_in,
                              void* d_out, int out_size, void* d_ws, size_t ws_size,
                              hipStream_t stream) {
  const float* x     = (const float*)d_in[0];
  const float* Wih_f = (const float*)d_in[1];
  const float* Whh_f = (const float*)d_in[2];
  const float* b_f   = (const float*)d_in[3];
  const float* Wih_r = (const float*)d_in[4];
  const float* Whh_r = (const float*)d_in[5];
  const float* b_r   = (const float*)d_in[6];
  /* d_in[7] Wih_d: unused by the reference */
  const float* Whh_d = (const float*)d_in[8];
  const float* b_d   = (const float*)d_in[9];
  const float* Wr_h  = (const float*)d_in[10];
  const float* br_h  = (const float*)d_in[11];
  const float* Wr_c  = (const float*)d_in[12];
  const float* br_c  = (const float*)d_in[13];
  const float* W1    = (const float*)d_in[14];
  const float* b1    = (const float*)d_in[15];
  const float* W2    = (const float*)d_in[16];
  const float* b2    = (const float*)d_in[17];
  const float* vt    = (const float*)d_in[18];
  const float* bvt   = (const float*)d_in[19];
  float* out = (float*)d_out;

  // ---- workspace layout -------------------------------------------------
  char* ws = (char*)d_ws;
  size_t off = 0;
  auto take = [&](size_t bytes) -> char* {
    char* p = ws + off;
    off += (bytes + 255) & ~(size_t)255;
    return p;
  };
  __bf16* xbf     = (__bf16*)take((size_t)Ss * Bb * Dd * 2);
  __bf16* wihf_bf = (__bf16*)take((size_t)HH4 * Dd * 2);
  __bf16* wihr_bf = (__bf16*)take((size_t)HH4 * Dd * 2);
  __bf16* whhf_bf = (__bf16*)take((size_t)HH4 * Hh * 2);
  __bf16* whhr_bf = (__bf16*)take((size_t)HH4 * Hh * 2);
  __bf16* whhd_bf = (__bf16*)take((size_t)HH4 * Hh * 2);
  __bf16* w1_bf   = (__bf16*)take((size_t)Hh * Hh * 2);
  __bf16* w2_bf   = (__bf16*)take((size_t)Hh * Hh * 2);
  __bf16* wrh_bf  = (__bf16*)take((size_t)Hh * 2 * Hh * 2);
  __bf16* wrc_bf  = (__bf16*)take((size_t)Hh * 2 * Hh * 2);
  __bf16* xprojf  = (__bf16*)take((size_t)Ss * Bb * HH4 * 2);   // 67 MB
  __bf16* xprojr  = (__bf16*)take((size_t)Ss * Bb * HH4 * 2);   // 67 MB
  float*  hs_sum  = (float*) take((size_t)Ss * Bb * Hh * 4);    // 33.5 MB
  float*  pre1    = (float*) take((size_t)Ss * Bb * Hh * 4);    // 33.5 MB
  __bf16* enc_bf  = (__bf16*)take((size_t)Ss * Bb * Hh * 2);    // 16.8 MB
  float*  h_f     = (float*) take((size_t)Bb * Hh * 4);
  float*  c_f     = (float*) take((size_t)Bb * Hh * 4);
  float*  h_r     = (float*) take((size_t)Bb * Hh * 4);
  float*  c_r     = (float*) take((size_t)Bb * Hh * 4);
  float*  h_d     = (float*) take((size_t)Bb * Hh * 4);
  float*  c_d     = (float*) take((size_t)Bb * Hh * 4);
  float*  qbuf    = (float*) take((size_t)Bb * Hh * 4);
  __bf16* hbf_f0  = (__bf16*)take((size_t)Bb * Hh * 2);
  __bf16* hbf_f1  = (__bf16*)take((size_t)Bb * Hh * 2);
  __bf16* hbf_r0  = (__bf16*)take((size_t)Bb * Hh * 2);
  __bf16* hbf_r1  = (__bf16*)take((size_t)Bb * Hh * 2);
  __bf16* hbf_d0  = (__bf16*)take((size_t)Bb * Hh * 2);
  __bf16* hbf_d1  = (__bf16*)take((size_t)Bb * Hh * 2);
  __bf16* hcat    = (__bf16*)take((size_t)Bb * 2 * Hh * 2);
  __bf16* hbf_f[2] = {hbf_f0, hbf_f1};
  __bf16* hbf_r[2] = {hbf_r0, hbf_r1};
  __bf16* hbf_d[2] = {hbf_d0, hbf_d1};
  (void)ws_size; (void)in_sizes; (void)n_in; (void)out_size;

  // ---- helpers ----------------------------------------------------------
  auto conv = [&](const float* in, __bf16* o, size_t n) {
    f2bf_kernel<<<(unsigned)((n + 255) / 256), 256, 0, stream>>>(in, o, n);
  };
  auto zero = [&](void* p, size_t bytes) {
    size_t nw = bytes >> 2;
    zero_kernel<<<(unsigned)((nw + 255) / 256), 256, 0, stream>>>((unsigned int*)p, nw);
  };
  auto gemm_bf = [&](const __bf16* A, const __bf16* W, const float* bias, __bf16* C,
                     int M, int N, int K) {
    int tiles = (M >> 5) * (N >> 5);
    wmma_gemm_bias<__bf16><<<(tiles + 3) / 4, 128, 0, stream>>>(A, W, bias, C, M, N, K);
  };
  auto gemm_f = [&](const __bf16* A, const __bf16* W, const float* bias, float* C,
                    int M, int N, int K) {
    int tiles = (M >> 5) * (N >> 5);
    wmma_gemm_bias<float><<<(tiles + 3) / 4, 128, 0, stream>>>(A, W, bias, C, M, N, K);
  };

  // ---- precision staging: fp32 -> bf16 ----------------------------------
  {
    size_t n = (size_t)Ss * Bb * Dd;
    x_transpose_bf<<<(unsigned)((n + 255) / 256), 256, 0, stream>>>(x, xbf);
  }
  conv(Wih_f, wihf_bf, (size_t)HH4 * Dd);
  conv(Wih_r, wihr_bf, (size_t)HH4 * Dd);
  conv(Whh_f, whhf_bf, (size_t)HH4 * Hh);
  conv(Whh_r, whhr_bf, (size_t)HH4 * Hh);
  conv(Whh_d, whhd_bf, (size_t)HH4 * Hh);
  conv(W1, w1_bf, (size_t)Hh * Hh);
  conv(W2, w2_bf, (size_t)Hh * Hh);
  conv(Wr_h, wrh_bf, (size_t)Hh * 2 * Hh);
  conv(Wr_c, wrc_bf, (size_t)Hh * 2 * Hh);

  // ---- big parallel GEMMs: x projections (bias folded in) ---------------
  gemm_bf(xbf, wihf_bf, b_f, xprojf, Ss * Bb, HH4, Dd);
  gemm_bf(xbf, wihr_bf, b_r, xprojr, Ss * Bb, HH4, Dd);

  // ---- init recurrent state ---------------------------------------------
  zero(h_f, (size_t)Bb * Hh * 4);
  zero(c_f, (size_t)Bb * Hh * 4);
  zero(h_r, (size_t)Bb * Hh * 4);
  zero(c_r, (size_t)Bb * Hh * 4);
  zero(hbf_f[0], (size_t)Bb * Hh * 2);
  zero(hbf_r[0], (size_t)Bb * Hh * 2);

  // ---- forward LSTM (writes hs_sum[s] = h_f(s)) -------------------------
  for (int s = 0; s < Ss; ++s) {
    lstm_step<<<Hh / 16, 64, 0, stream>>>(
        hbf_f[s & 1], whhf_bf, xprojf + (size_t)s * Bb * HH4, nullptr,
        h_f, c_f, hbf_f[(s + 1) & 1], hs_sum + (size_t)s * Bb * Hh, 0);
  }
  // ---- reverse LSTM (accumulates hs_sum[s] += h_r at original index) ----
  for (int tt = 0; tt < Ss; ++tt) {
    const int s = Ss - 1 - tt;
    lstm_step<<<Hh / 16, 64, 0, stream>>>(
        hbf_r[tt & 1], whhr_bf, xprojr + (size_t)s * Bb * HH4, nullptr,
        h_r, c_r, hbf_r[(tt + 1) & 1], hs_sum + (size_t)s * Bb * Hh, 1);
  }

  // ---- decoder init: h_d/c_d = [hf,hr] @ Wr^T + br ----------------------
  concat_bf<<<(Bb * 2 * Hh + 255) / 256, 256, 0, stream>>>(h_f, h_r, hcat);
  gemm_f(hcat, wrh_bf, br_h, h_d, Bb, Hh, 2 * Hh);
  gemm_f(hcat, wrc_bf, br_c, c_d, Bb, Hh, 2 * Hh);

  // ---- pre1 = enc_red @ W1^T + b1 (L2-resident for decoder) -------------
  conv(hs_sum, enc_bf, (size_t)Ss * Bb * Hh);
  gemm_f(enc_bf, w1_bf, b1, pre1, Ss * Bb, Hh, Hh);
  conv(h_d, hbf_d[0], (size_t)Bb * Hh);

  // ---- decoder loop ------------------------------------------------------
  for (int t = 0; t < Ss; ++t) {
    lstm_step<<<Hh / 16, 64, 0, stream>>>(
        hbf_d[t & 1], whhd_bf, nullptr, b_d,
        h_d, c_d, hbf_d[(t + 1) & 1], nullptr, 0);
    gemm_f(hbf_d[(t + 1) & 1], w2_bf, b2, qbuf, Bb, Hh, Hh);
    attn_scores<<<(Ss * Bb) / 8, 256, 0, stream>>>(pre1, qbuf, vt, bvt, out, t);
  }
}
